// AdaptiveSparseAttention_74577812127865
// MI455X (gfx1250) — matrix operand order repaired and checked by reading, not compiled, not run on hardware
//
#include <hip/hip_runtime.h>

// ---------------------------------------------------------------------------
// Types for CDNA5 WMMA / TDM builtins
// ---------------------------------------------------------------------------
typedef __attribute__((ext_vector_type(16))) __bf16        v16bf;
typedef __attribute__((ext_vector_type(8)))  float         v8f;
typedef __attribute__((ext_vector_type(4)))  unsigned int  v4u;
typedef __attribute__((ext_vector_type(8)))  int           v8i;
typedef __attribute__((ext_vector_type(4)))  int           v4i;

#define NEGBIG (-1e30f)

// ---------------------------------------------------------------------------
// Helpers
// ---------------------------------------------------------------------------
__device__ inline v8f vzero8() {
    v8f z;
#pragma unroll
    for (int i = 0; i < 8; ++i) z[i] = 0.f;
    return z;
}

__device__ inline v8f wmma_bf16(v16bf a, v16bf b, v8f c) {
    // D = A(16x32 bf16) * B(32x16 bf16) + C(16x16 f32)
    return __builtin_amdgcn_wmma_f32_16x16x32_bf16(
        false, a, false, b, (short)0, c, false, false);
}

// The TDM builtin consumes descriptor VALUES, so LLVM cannot see that it
// writes LDS. Escape the LDS pointer through an asm with a memory clobber so
// loads after this point are real ds_reads (not folded to undef / hoisted).
__device__ inline void publish_lds(void* p) {
    asm volatile("" : "+v"(p) :: "memory");
}

// A-matrix fragment (16x32 bf16): lane holds row (lane&15);
// element e -> K = hi*16 + e (matches B fragments).
__device__ inline v16bf make_a_frag(const float* rowBase, int hi) {
    v16bf a;
    const float* p0 = rowBase + hi * 16;
    const float* p1 = rowBase + hi * 16 + 8;
#pragma unroll
    for (int e = 0; e < 8; ++e) a[e] = (__bf16)p0[e];
#pragma unroll
    for (int e = 0; e < 8; ++e) a[8 + e] = (__bf16)p1[e];
    return a;
}

// B-matrix fragment (32x16 bf16): lane holds column (lane&15); element e has
// K = hi*16 + e. "base" points at element (K = 0, N = col); row stride
// strideF floats.
__device__ inline v16bf make_b_frag_strided(const float* base, int strideF, int hi) {
    v16bf b;
    const float* p = base + (size_t)(hi * 16) * strideF;
#pragma unroll
    for (int e = 0; e < 16; ++e) b[e] = (__bf16)p[(size_t)e * strideF];
    return b;
}

// B fragment where K runs contiguously in memory (caller folds in hi*16).
__device__ inline v16bf make_b_frag_contig(const float* baseWithHi) {
    v16bf b;
#pragma unroll
    for (int e = 0; e < 16; ++e) b[e] = (__bf16)baseWithHi[e];
    return b;
}

__device__ inline float wredMax(float v) {
#pragma unroll
    for (int o = 16; o > 0; o >>= 1) v = fmaxf(v, __shfl_xor(v, o, 32));
    return v;
}
__device__ inline float wredSum(float v) {
#pragma unroll
    for (int o = 16; o > 0; o >>= 1) v += __shfl_xor(v, o, 32);
    return v;
}

// Monotone float -> u32 key (order-preserving, handles negatives).
__device__ inline unsigned fkey(float f) {
    unsigned u = __float_as_uint(f);
    return (u & 0x80000000u) ? ~u : (u | 0x80000000u);
}

// ---------------------------------------------------------------------------
// Tensor Data Mover: 2D tile (f32) global -> LDS.  D# per ISA cdna5 ch.8:
// group0 = {count/lds_addr/global_addr/type}, group1 = dims/strides.
// 6-arg builtin on this toolchain: (v4u, v8i, v4i, v4i, v8i, i32 cpol).
// ---------------------------------------------------------------------------
__device__ inline void tdm_load_2d_f32(unsigned ldsByteOff, const void* gptr,
                                       unsigned tileW, unsigned tileH,
                                       unsigned strideElems) {
    unsigned long long ga = (unsigned long long)(size_t)gptr;
    v4u g0;
    g0.x = 1u;                                   // count=1, user mode, no gather
    g0.y = ldsByteOff;                           // lds_addr
    g0.z = (unsigned)ga;                         // global_addr[31:0]
    g0.w = (unsigned)((ga >> 32) & 0x1FFFFFFull) | (2u << 30);  // addr[56:32], type=2
    unsigned td0 = strideElems;                  // tensor_dim0 (row length)
    unsigned td1 = 0x7FFFFFFFu;                  // tensor_dim1 (large, tiles in-bounds)
    v8i g1;
    g1[0] = (int)(2u << 16);                                   // data_size=2 (4B)
    g1[1] = (int)((td0 & 0xFFFFu) << 16);                      // tensor_dim0 lo16
    g1[2] = (int)((td0 >> 16) | ((td1 & 0xFFFFu) << 16));      // dim0 hi16 | dim1 lo16
    g1[3] = (int)((td1 >> 16) | (tileW << 16));                // dim1 hi16 | tile_dim0
    g1[4] = (int)(tileH & 0xFFFFu);                            // tile_dim1 (tile_dim2=0)
    g1[5] = (int)strideElems;                                  // tensor_dim0_stride lo32
    g1[6] = 0;                                                 // stride hi16 | dim1_stride lo16
    g1[7] = 0;
    v4i z4 = {0, 0, 0, 0};
    v8i z8 = {0, 0, 0, 0, 0, 0, 0, 0};
    __builtin_amdgcn_tensor_load_to_lds(g0, g1, z4, z4, z8, 0);
}

// ---------------------------------------------------------------------------
// Kernel 1/3: Out[M,N] = A[M,K] * W[K,N] + bias[N]   (bf16 WMMA, f32 acc)
// Block = 256 thr (8 waves), tile 128x128, K step 32.
// Double-buffered TDM pipeline: issue next tiles, s_wait_tensorcnt(2) so the
// two in-flight loads overlap the current tile's WMMAs.
// ---------------------------------------------------------------------------
#define TILEA (128 * 32)
#define TILEB (32 * 128)
#define TILE  (TILEA + TILEB)

__global__ void __launch_bounds__(256)
gemm_bias_kernel(const float* __restrict__ A, const float* __restrict__ W,
                 const float* __restrict__ bias, float* __restrict__ Out,
                 int M, int N, int K) {
    __shared__ float lds[2 * TILE];              // 2 x (A[128][32] + B[32][128])

    const int tid = threadIdx.x;
    const int w = tid >> 5, lane = tid & 31;
    const int lane16 = lane & 15, hi = lane >> 4;
    const int mBase = blockIdx.y * 128, nBase = blockIdx.x * 128;
    const int wm = (w & 3) * 32;      // 4 M-slices of 32
    const int wn = (w >> 2) * 64;     // 2 N-slices of 64

    v8f acc[2][4];
#pragma unroll
    for (int mt = 0; mt < 2; ++mt)
#pragma unroll
        for (int nt = 0; nt < 4; ++nt) acc[mt][nt] = vzero8();

    const int nk = K >> 5;
    if (tid == 0) {
        tdm_load_2d_f32(0u, A + (size_t)mBase * K, 32u, 128u, (unsigned)K);
        tdm_load_2d_f32((unsigned)(TILEA * 4), W + nBase, 128u, 32u, (unsigned)N);
    }

    for (int i = 0; i < nk; ++i) {
        const int buf = i & 1;
        float* ldsA = lds + buf * TILE;
        float* ldsB = ldsA + TILEA;

        __syncthreads();   // all waves done reading buf^1 (previous iteration)
        if (tid == 0) {
            if (i + 1 < nk) {
                const int k1 = (i + 1) * 32;
                const unsigned off = (unsigned)((buf ^ 1) * TILE * 4);
                tdm_load_2d_f32(off, A + (size_t)mBase * K + k1, 32u, 128u, (unsigned)K);
                tdm_load_2d_f32(off + (unsigned)(TILEA * 4), W + (size_t)k1 * N + nBase,
                                128u, 32u, (unsigned)N);
                __builtin_amdgcn_s_wait_tensorcnt(2);   // current buf complete
            } else {
                __builtin_amdgcn_s_wait_tensorcnt(0);
            }
        }
        __syncthreads();
        publish_lds(lds);  // TDM wrote LDS behind the compiler's back

        v16bf afr[2];
        afr[0] = make_a_frag(ldsA + (wm + 0  + lane16) * 32, hi);
        afr[1] = make_a_frag(ldsA + (wm + 16 + lane16) * 32, hi);
#pragma unroll
        for (int nt = 0; nt < 4; ++nt) {
            v16bf bfr = make_b_frag_strided(ldsB + wn + nt * 16 + lane16, 128, hi);
            acc[0][nt] = wmma_bf16(afr[0], bfr, acc[0][nt]);
            acc[1][nt] = wmma_bf16(afr[1], bfr, acc[1][nt]);
        }
    }

#pragma unroll
    for (int mt = 0; mt < 2; ++mt)
#pragma unroll
        for (int nt = 0; nt < 4; ++nt)
#pragma unroll
            for (int r = 0; r < 8; ++r) {
                int row = mBase + wm + mt * 16 + hi * 8 + r;
                int col = nBase + wn + nt * 16 + lane16;
                Out[(size_t)row * N + col] = acc[mt][nt][r] + bias[col];
            }
}

// ---------------------------------------------------------------------------
// Kernel 2: fused adaptive sparse attention, one block per (b,h,16 q-rows).
// Never materializes the [B,nH,T,T] att tensor.
// ---------------------------------------------------------------------------
#define TT 1024
#define HD 64

__global__ void __launch_bounds__(256)
attn_kernel(const float* __restrict__ qkv, const float* __restrict__ noise,
            const float* __restrict__ sr, const float* __restrict__ gt,
            float* __restrict__ yout, int B, int T, int C) {
    extern __shared__ float smem[];
    float*    S    = smem;                 // [16][1024] scores -> sparse
    float*    P    = S + 16 * TT;          // [16][1024] gumbel logits -> p
    float*    Qt   = P + 16 * TT;          // [16][64]
    unsigned* hist = (unsigned*)(Qt + 16 * HD);  // [16][256]
    float*    Yp   = (float*)(hist + 16 * 256);  // [16][64] partial y

    const int tid = threadIdx.x, w = tid >> 5, lane = tid & 31;
    const int lane16 = lane & 15, hi = lane >> 4;
    const int qBase = blockIdx.x * 16;
    const int bh = blockIdx.y, b = bh >> 4, h = bh & 15;
    const int C3 = 3 * C;

    const float ratio    = 1.f / (1.f + __expf(-sr[h]));
    const float inv_temp = 1.f / (log1pf(__expf(gt[h])) + 0.1f);
    const float scale    = 0.125f;   // 1/sqrt(64)

    // ---- stage Q tile (f32) -------------------------------------------------
    for (int i = tid; i < 16 * HD; i += 256) {
        int r = i >> 6, d = i & 63;
        Qt[i] = qkv[(size_t)(b * T + qBase + r) * C3 + h * HD + d];
    }
    __syncthreads();

    // ---- scores S[16][T] via WMMA: wave w -> j-tiles w, w+8, ..., w+56 ------
    const float* Kbase = qkv + (size_t)b * T * C3 + C + h * HD;
    for (int s = 0; s < 8; ++s) {
        int j0 = (w + 8 * s) * 16;
        v8f acc = vzero8();
#pragma unroll
        for (int c = 0; c < 2; ++c) {   // hd chunks of 32
            v16bf af = make_a_frag(Qt + lane16 * HD + c * 32, hi);
            const float* kb = Kbase + (size_t)(j0 + lane16) * C3 + c * 32 + hi * 16;
            v16bf bf = make_b_frag_contig(kb);   // K^T: hd runs contiguous
            acc = wmma_bf16(af, bf, acc);
        }
#pragma unroll
        for (int r = 0; r < 8; ++r)
            S[(hi * 8 + r) * TT + j0 + lane16] = acc[r] * scale;
    }
    __syncthreads();

    // ---- gumbel logits; mask to causal --------------------------------------
    const float* NU = noise + ((size_t)bh * T + qBase) * TT;
    for (int i = tid; i < 16 * TT; i += 256) {
        int r = i >> 10, j = i & (TT - 1);
        int t = qBase + r;
        float sc = S[i];
        float u  = NU[(size_t)r * TT + j];
        float g  = -__logf(-__logf(u + 1e-8f) + 1e-8f);
        bool causal = (j <= t);
        P[i] = causal ? (sc + g) * inv_temp : NEGBIG;
        S[i] = causal ? sc : 0.f;
    }
    __syncthreads();

    // ---- per-row exact top-k (radix select) + two softmaxes -----------------
    // Wave w owns rows 2w, 2w+1; all selection state is wave-local.
    for (int rr = 0; rr < 2; ++rr) {
        const int r = w * 2 + rr;
        const int t = qBase + r;
        const int nvalid = t + 1;
        int ksel = (int)floorf((float)nvalid * ratio);
        if (ksel < 1) ksel = 1;

        float* GLrow = P + r * TT;
        float* Srow  = S + r * TT;
        unsigned* hrow = hist + r * 256;

        unsigned thr = 0u;   // all-selected default
        if (ksel < nvalid) {
            unsigned prefix = 0u;
            int kneed = ksel;
            for (int pass = 3; pass >= 0; --pass) {
                const int shift = pass * 8;
                const unsigned maskHi = (pass == 3) ? 0u : (0xFFFFFFFFu << (shift + 8));
                for (int i2 = lane; i2 < 256; i2 += 32) hrow[i2] = 0u;
                __threadfence_block();
                for (int j = lane; j < nvalid; j += 32) {
                    unsigned key = fkey(GLrow[j]);
                    if ((key & maskHi) == (prefix & maskHi))
                        atomicAdd(&hrow[(key >> shift) & 255u], 1u);
                }
                __threadfence_block();
                int cum = 0, chosen = 0, kn2 = kneed;
                bool found = false;
                for (int bbin = 255; bbin >= 0; --bbin) {
                    int c = (int)hrow[bbin];
                    if (!found && cum + c >= kneed) { chosen = bbin; kn2 = kneed - cum; found = true; }
                    cum += c;
                }
                prefix |= ((unsigned)chosen) << shift;
                kneed = kn2;
            }
            thr = prefix;   // exact key of k-th largest (ties measure-zero)
        }

        // softmax of gumbel logits over the selected set
        float m1 = -3.4e38f;
        for (int j = lane; j < nvalid; j += 32) {
            float gl = GLrow[j];
            if (fkey(gl) >= thr) m1 = fmaxf(m1, gl);
        }
        m1 = wredMax(m1);
        float s1 = 0.f;
        for (int j = lane; j < nvalid; j += 32) {
            float gl = GLrow[j];
            if (fkey(gl) >= thr) s1 += __expf(gl - m1);
        }
        s1 = wredSum(s1);
        const float inv_s1 = 1.f / s1;

        // sparse = selected ? att_safe * sv : 0   (full row, incl. j>t -> 0)
        for (int j = lane; j < TT; j += 32) {
            float v = 0.f;
            if (j < nvalid) {
                float gl = GLrow[j];
                if (fkey(gl) >= thr) v = Srow[j] * (__expf(gl - m1) * inv_s1);
            }
            Srow[j] = v;
        }
        __threadfence_block();

        // full-row softmax (zeros at unselected positions contribute exp(0))
        float m2 = 0.f;
        for (int j = lane; j < TT; j += 32) m2 = fmaxf(m2, Srow[j]);
        m2 = wredMax(m2);
        float s2 = 0.f;
        for (int j = lane; j < TT; j += 32) s2 += __expf(Srow[j] - m2);
        s2 = wredSum(s2);
        const float inv_s2 = 1.f / s2;
        for (int j = lane; j < TT; j += 32)
            GLrow[j] = __expf(Srow[j] - m2) * inv_s2;   // P now holds p
        __threadfence_block();
    }
    __syncthreads();

    // ---- y = p @ V via WMMA: wave = (khalf, ntile); K split in halves -------
    const float* Vbase = qkv + (size_t)b * T * C3 + 2 * C + h * HD;
    const int khalf = w >> 2, nt = w & 3;
    v8f acc = vzero8();
    for (int kc = khalf * 16; kc < khalf * 16 + 16; ++kc) {
        int kb = kc * 32;
        v16bf af = make_a_frag(P + lane16 * TT + kb, hi);
        v16bf bf = make_b_frag_strided(Vbase + (size_t)kb * C3 + nt * 16 + lane16, C3, hi);
        acc = wmma_bf16(af, bf, acc);
    }
    if (khalf == 1) {
#pragma unroll
        for (int r = 0; r < 8; ++r)
            Yp[(hi * 8 + r) * HD + nt * 16 + lane16] = acc[r];
    }
    __syncthreads();
    if (khalf == 0) {
#pragma unroll
        for (int r = 0; r < 8; ++r) {
            int m = hi * 8 + r, col = nt * 16 + lane16;
            float val = acc[r] + Yp[m * HD + col];
            yout[(size_t)(b * T + qBase + m) * C + h * HD + col] = val;
        }
    }
}

// ---------------------------------------------------------------------------
// Host launcher
// ---------------------------------------------------------------------------
extern "C" void kernel_launch(void* const* d_in, const int* in_sizes, int n_in,
                              void* d_out, int out_size, void* d_ws, size_t ws_size,
                              hipStream_t stream) {
    (void)in_sizes; (void)n_in; (void)out_size; (void)ws_size;
    const float* x      = (const float*)d_in[0];
    const float* W_attn = (const float*)d_in[1];
    const float* b_attn = (const float*)d_in[2];
    const float* W_proj = (const float*)d_in[3];
    const float* b_proj = (const float*)d_in[4];
    const float* sr     = (const float*)d_in[5];
    const float* gt     = (const float*)d_in[6];
    const float* nu     = (const float*)d_in[7];
    float* out = (float*)d_out;

    const int Bb = 4, T = 1024, C = 1024;
    float* qkv = (float*)d_ws;                       // [4096, 3072]
    float* y   = qkv + (size_t)Bb * T * 3 * C;       // [4096, 1024]

    dim3 blk(256);

    // qkv = x @ W_attn + b_attn
    gemm_bias_kernel<<<dim3((3 * C) / 128, (Bb * T) / 128), blk, 0, stream>>>(
        x, W_attn, b_attn, qkv, Bb * T, 3 * C, C);

    // fused adaptive sparse attention -> y
    size_t smemAtt = (size_t)(16 * TT + 16 * TT + 16 * HD + 16 * HD) * sizeof(float)
                   + (size_t)(16 * 256) * sizeof(unsigned);
    (void)hipFuncSetAttribute((const void*)attn_kernel,
                              hipFuncAttributeMaxDynamicSharedMemorySize,
                              (int)smemAtt);
    attn_kernel<<<dim3(T / 16, Bb * 16), blk, smemAtt, stream>>>(
        qkv, nu, sr, gt, y, Bb, T, C);

    // out = y @ W_proj + b_proj
    gemm_bias_kernel<<<dim3(C / 128, (Bb * T) / 128), blk, 0, stream>>>(
        y, W_proj, b_proj, out, Bb * T, C, C);
}
